// ProteinAtomicChimeraDenseEmbedder_56023553409098
// MI455X (gfx1250) — matrix operand order repaired and checked by reading, not compiled, not run on hardware
//
#include <hip/hip_runtime.h>
#include <math.h>

// ----------------------------- model dims -----------------------------------
#define NATOM   5376
#define ADIM    56
#define EBOND   10752
#define ERAD    86016
#define ETOT    96768
#define NRES    384
#define CS      128
#define CZ      128
#define NHD     16
#define PQK     8
#define PVV     12
#define RDIM    112
#define CATD    3072
#define NPAIR   (NRES*NRES)

typedef __bf16 bf16;
typedef __attribute__((ext_vector_type(16))) __bf16 v16bf;
typedef __attribute__((ext_vector_type(8)))  float  v8f;

// CDNA5 async global->LDS path (ASYNCcnt). Guarded so absence degrades to the
// plain cooperative copy instead of failing the build.
#if defined(__gfx1250__) && __has_builtin(__builtin_amdgcn_global_load_async_to_lds_b128) && \
    __has_builtin(__builtin_amdgcn_s_wait_asynccnt)
#define USE_ASYNC_LDS 1
// Builtin prototype (from hipcc diagnostic): (v4i AS1*, v4i AS3*, imm offset, imm cpol)
typedef int async_v4i __attribute__((vector_size(16)));
typedef __attribute__((address_space(1))) async_v4i* async_gptr;
typedef __attribute__((address_space(3))) async_v4i* async_lptr;
#else
#define USE_ASYNC_LDS 0
#endif

union FragU { uint4 q[2]; v16bf v; };

__device__ __forceinline__ v8f wmma_bf(v16bf a, v16bf b, v8f c) {
  return __builtin_amdgcn_wmma_f32_16x16x32_bf16(false, a, false, b, (short)0, c, false, false);
}

// A fragment (16x32 bf16, row-major source, row stride lda elems).
// lane<16: row=lane, K k0+0..15 ; lane>=16: row=lane-16, K k0+16..31.
__device__ __forceinline__ v16bf load_a_frag(const bf16* base, int lda, int k0, int lane) {
  int row = lane & 15;
  int off = (lane < 16) ? 0 : 16;
  const bf16* p = base + (size_t)row * lda + k0 + off;
  FragU u;
  u.q[0] = *(const uint4*)(p);
  u.q[1] = *(const uint4*)(p + 8);
  return u.v;
}
// Packed-B fragment: 16 bf16 contiguous per lane.
__device__ __forceinline__ v16bf load_b_frag(const bf16* pb, int lane) {
  FragU u;
  const bf16* p = pb + (size_t)lane * 16;
  u.q[0] = *(const uint4*)(p);
  u.q[1] = *(const uint4*)(p + 8);
  return u.v;
}

// ----------------------- weight packing (B matrices) -------------------------
// Packed layout: ((kt*ntiles + nt)*32 + lane)*16 + e ; element = B[(kt*32 + (lane<16?0:16) + e), nt*16 + lane&15]
__global__ void pack_b_kernel(const float* __restrict__ B, bf16* __restrict__ out,
                              int Ksrc, int Kpad, int N) {
  int t = blockIdx.x * 256 + threadIdx.x;
  int total = (Kpad >> 5) * (N >> 4) * 512;
  if (t >= total) return;
  int e = t & 15, lane = (t >> 4) & 31, ft = t >> 9;
  int ntiles = N >> 4;
  int nt = ft % ntiles, kt = ft / ntiles;
  int kk = kt * 32 + ((lane < 16) ? 0 : 16) + e;
  int n  = nt * 16 + (lane & 15);
  float v = (kk < Ksrc) ? B[(size_t)kk * N + n] : 0.f;
  out[t] = (bf16)v;
}

// Attention-apply B: per head h, K=384(j), N=64: cols[0..15]=v[j,h,c], [16..51]=vp_glob[j,h,*], [52..63]=0
__global__ void pack_apply_b(const float* __restrict__ vbuf, const float* __restrict__ vpg,
                             bf16* __restrict__ out) {
  int t = blockIdx.x * 256 + threadIdx.x;
  if (t >= 16 * 12 * 4 * 512) return;
  int e = t & 15, lane = (t >> 4) & 31, r = t >> 9;
  int nt = r & 3; r >>= 2;
  int kt = r % 12, h = r / 12;
  int kj = kt * 32 + ((lane < 16) ? 0 : 16) + e;
  int n  = nt * 16 + (lane & 15);
  float v = 0.f;
  if (n < 16)      v = vbuf[(size_t)kj * 256 + h * 16 + n];
  else if (n < 52) v = vpg[(size_t)kj * 576 + h * 36 + (n - 16)];
  out[t] = (bf16)v;
}

// ------------------------- generic WMMA GEMM --------------------------------
// C[M,N] (f32 and/or bf16) = A[M,K](bf16,row-major,lda) x Bpacked + bias ; optional relu.
// 128 threads = 4 waves, one 16x16 tile per wave. Optional batch via blockIdx.y.
__global__ void gemm_bf16_wmma(const bf16* __restrict__ A, int lda, const bf16* __restrict__ Bp,
                               const float* __restrict__ bias, float* __restrict__ C,
                               bf16* __restrict__ Cb, int M, int N, int K, int relu,
                               long long sA, long long sB, long long sC) {
  int batch = blockIdx.y;
  A  += (size_t)batch * sA;
  Bp += (size_t)batch * sB;
  size_t coff = (size_t)batch * sC;
  int lane = threadIdx.x & 31;
  int tile = blockIdx.x * 4 + (threadIdx.x >> 5);
  int ntiles = N >> 4, mtiles = M >> 4;
  if (tile >= mtiles * ntiles) return;
  int mt = tile / ntiles, nt = tile % ntiles;
  v8f acc = {};
  const bf16* Abase = A + (size_t)(mt * 16) * lda;
  int ksteps = K >> 5;
  for (int kt = 0; kt < ksteps; ++kt) {
    v16bf av = load_a_frag(Abase, lda, kt * 32, lane);
    v16bf bv = load_b_frag(Bp + (size_t)(kt * ntiles + nt) * 512, lane);
    acc = wmma_bf(av, bv, acc);
  }
  int col = nt * 16 + (lane & 15);
  float bs = bias ? bias[col] : 0.f;
  for (int r = 0; r < 8; ++r) {
    int row = mt * 16 + r + ((lane >> 4) << 3);
    float v = acc[r] + bs;
    if (relu) v = fmaxf(v, 0.f);
    if (C)  C[coff + (size_t)row * N + col] = v;
    if (Cb) Cb[coff + (size_t)row * N + col] = (bf16)v;
  }
}

// ------------------------------ small kernels -------------------------------
__global__ void f32_to_bf16_kernel(const float* __restrict__ in, bf16* __restrict__ out, long long n) {
  long long i = (long long)blockIdx.x * 256 + threadIdx.x;
  if (i < n) out[i] = (bf16)in[i];
}

__global__ void edge_mlp_kernel(const float* __restrict__ feat, const float* __restrict__ W,
                                const float* __restrict__ b, float* __restrict__ out, int E) {
  __shared__ float Ws[256]; __shared__ float bs[16];
  int t = threadIdx.x;
  if (t < 256) Ws[t] = W[t];
  if (t < 16)  bs[t] = b[t];
  __syncthreads();
  int e = blockIdx.x * 256 + t;
  if (e >= E) return;
  float f[16];
  for (int c = 0; c < 16; ++c) f[c] = feat[(size_t)e * 16 + c];
  for (int o = 0; o < 16; ++o) {
    float acc = bs[o];
    for (int c = 0; c < 16; ++c) acc += f[c] * Ws[c * 16 + o];
    out[(size_t)e * 16 + o] = fmaxf(acc, 0.f);
  }
}

__global__ void quat_rot_kernel(const float* __restrict__ quats, float* __restrict__ R) {
  int n = blockIdx.x * 256 + threadIdx.x;
  if (n >= NRES) return;
  const float* q = quats + n * 4;
  float w = q[0], x = q[1], y = q[2], z = q[3];
  float inv = rsqrtf(w * w + x * x + y * y + z * z);
  w *= inv; x *= inv; y *= inv; z *= inv;
  float* r = R + n * 9;
  r[0] = 1.f - 2.f * (y * y + z * z); r[1] = 2.f * (x * y - w * z); r[2] = 2.f * (x * z + w * y);
  r[3] = 2.f * (x * y + w * z); r[4] = 1.f - 2.f * (x * x + z * z); r[5] = 2.f * (y * z - w * x);
  r[6] = 2.f * (x * z - w * y); r[7] = 2.f * (y * z + w * x); r[8] = 1.f - 2.f * (x * x + y * y);
}

// ----------------- tensor-product messages + scatter-mean -------------------
template <int COS, int COV>
__global__ void tp_msg_kernel(const float* __restrict__ x, const float* __restrict__ sh,
                              const float* __restrict__ hb, const int* __restrict__ src,
                              const int* __restrict__ dst,
                              const float* w_ss, const float* w_sv, const float* w_vs,
                              const float* w_vv_s, const float* w_vv_x, const float* w_ve_s,
                              const float* w_ve_dot, const float* w_ve_x, const float* w_so_s,
                              const float* w_so_v,
                              float* __restrict__ sums, float* __restrict__ cnt, int E) {
  extern __shared__ float L[];
  const float* gsrc[10] = {w_ss, w_sv, w_vs, w_vv_s, w_vv_x, w_ve_s, w_ve_dot, w_ve_x, w_so_s, w_so_v};
  const int gsz[10] = {256 * COS, 256 * COV, 64 * COV, 64 * COS, 64 * COV,
                       64 * COV, 64 * COS, 64 * COV, 256 * COS, 256 * COV};
  float* seg[10];
  int base = 0;
  for (int s = 0; s < 10; ++s) {
    seg[s] = L + base;
    for (int k = threadIdx.x; k < gsz[s]; k += blockDim.x) seg[s][k] = gsrc[s][k];
    base += gsz[s];
  }
  __syncthreads();
  const float *Lss = seg[0], *Lsv = seg[1], *Lvs = seg[2], *Lvvs = seg[3], *Lvvx = seg[4],
              *Lves = seg[5], *Lved = seg[6], *Lvex = seg[7], *Lsos = seg[8], *Lsov = seg[9];

  int e = blockIdx.x * blockDim.x + threadIdx.x;
  if (e >= E) return;
  float th_[16];
  for (int h = 0; h < 16; ++h) th_[h] = hb[(size_t)e * 16 + h];
  float sh0 = sh[(size_t)e * 4];
  float s1[3] = {sh[(size_t)e * 4 + 1], sh[(size_t)e * 4 + 2], sh[(size_t)e * 4 + 3]};
  int sidx = src ? src[e] : e;
  const float* xr = x + (size_t)sidx * ADIM;
  float se_[16], so_[16], vo[4][3], ve[4][3];
  for (int i = 0; i < 16; ++i) { se_[i] = xr[i]; so_[i] = xr[40 + i]; }
  for (int i = 0; i < 4; ++i)
    for (int c = 0; c < 3; ++c) { vo[i][c] = xr[16 + i * 3 + c]; ve[i][c] = xr[28 + i * 3 + c]; }
  float dvo[4], dve[4], cvo[4][3], cve[4][3];
  for (int i = 0; i < 4; ++i) {
    dvo[i] = vo[i][0] * s1[0] + vo[i][1] * s1[1] + vo[i][2] * s1[2];
    dve[i] = ve[i][0] * s1[0] + ve[i][1] * s1[1] + ve[i][2] * s1[2];
    cvo[i][0] = vo[i][1] * s1[2] - vo[i][2] * s1[1];
    cvo[i][1] = vo[i][2] * s1[0] - vo[i][0] * s1[2];
    cvo[i][2] = vo[i][0] * s1[1] - vo[i][1] * s1[0];
    cve[i][0] = ve[i][1] * s1[2] - ve[i][2] * s1[1];
    cve[i][1] = ve[i][2] * s1[0] - ve[i][0] * s1[2];
    cve[i][2] = ve[i][0] * s1[1] - ve[i][1] * s1[0];
  }
  int d = dst[e];
  float* srow = sums + (size_t)d * (2 * COS + 6 * COV);

  { // pass A: m0e, m0o
    float m0e[COS] = {}, a0e[COS] = {}, m0o[COS] = {}, a0o[COS] = {};
    for (int h = 0; h < 16; ++h) {
      float th = th_[h];
      const float* pss = Lss + h * 16 * COS;
      const float* pso = Lsos + h * 16 * COS;
      for (int i = 0; i < 16; ++i) {
        float f = th * se_[i], g2 = th * so_[i];
        for (int o = 0; o < COS; ++o) { m0e[o] += f * pss[i * COS + o]; m0o[o] += g2 * pso[i * COS + o]; }
      }
      const float* pvv = Lvvs + h * 4 * COS;
      const float* pvd = Lved + h * 4 * COS;
      for (int i = 0; i < 4; ++i) {
        float a1 = th * dvo[i], a2 = th * dve[i];
        for (int o = 0; o < COS; ++o) { a0e[o] += a1 * pvv[i * COS + o]; a0o[o] += a2 * pvd[i * COS + o]; }
      }
    }
    for (int o = 0; o < COS; ++o) {
      atomicAdd(&srow[o], sh0 * m0e[o] + a0e[o]);
      atomicAdd(&srow[COS + 6 * COV + o], sh0 * m0o[o] + a0o[o]);
    }
  }
  { // pass B: m1o
    float sva[COV] = {}, vvo[COV][3] = {}, vce[COV][3] = {};
    for (int h = 0; h < 16; ++h) {
      float th = th_[h];
      const float* psv = Lsv + h * 16 * COV;
      for (int i = 0; i < 16; ++i) {
        float f = th * se_[i];
        for (int o = 0; o < COV; ++o) sva[o] += f * psv[i * COV + o];
      }
      const float* pvs = Lvs + h * 4 * COV;
      const float* pvx = Lvex + h * 4 * COV;
      for (int i = 0; i < 4; ++i)
        for (int c = 0; c < 3; ++c) {
          float b1 = th * vo[i][c], b4 = th * cve[i][c];
          for (int o = 0; o < COV; ++o) { vvo[o][c] += b1 * pvs[i * COV + o]; vce[o][c] += b4 * pvx[i * COV + o]; }
        }
    }
    for (int o = 0; o < COV; ++o)
      for (int c = 0; c < 3; ++c)
        atomicAdd(&srow[COS + o * 3 + c], s1[c] * sva[o] + sh0 * vvo[o][c] + vce[o][c]);
  }
  { // pass C: m1e
    float sov[COV] = {}, vve[COV][3] = {}, vco[COV][3] = {};
    for (int h = 0; h < 16; ++h) {
      float th = th_[h];
      const float* pov = Lsov + h * 16 * COV;
      for (int i = 0; i < 16; ++i) {
        float g2 = th * so_[i];
        for (int o = 0; o < COV; ++o) sov[o] += g2 * pov[i * COV + o];
      }
      const float* pes = Lves + h * 4 * COV;
      const float* pvx = Lvvx + h * 4 * COV;
      for (int i = 0; i < 4; ++i)
        for (int c = 0; c < 3; ++c) {
          float b2 = th * ve[i][c], b3 = th * cvo[i][c];
          for (int o = 0; o < COV; ++o) { vve[o][c] += b2 * pes[i * COV + o]; vco[o][c] += b3 * pvx[i * COV + o]; }
        }
    }
    for (int o = 0; o < COV; ++o)
      for (int c = 0; c < 3; ++c)
        atomicAdd(&srow[COS + 3 * COV + o * 3 + c], sh0 * vve[o][c] + vco[o][c] + s1[c] * sov[o]);
  }
  atomicAdd(&cnt[d], 1.f);
}

__global__ void finalize_atom_kernel(const float* __restrict__ af, const float* __restrict__ sums,
                                     const float* __restrict__ cnt, float* __restrict__ out) {
  int idx = blockIdx.x * 256 + threadIdx.x;
  if (idx >= NATOM * ADIM) return;
  int a = idx / ADIM;
  out[idx] = af[idx] + sums[idx] / fmaxf(cnt[a], 1.f);
}
__global__ void finalize_res_kernel(const float* __restrict__ sums, const float* __restrict__ cnt,
                                    float* __restrict__ out) {
  int idx = blockIdx.x * 256 + threadIdx.x;
  if (idx >= NRES * RDIM) return;
  int n = idx / RDIM;
  out[idx] = sums[idx] / fmaxf(cnt[n], 1.f);
}

// rotate vector channels by R^T, emit bf16 row padded to 128 (cols 112..127 = 0)
__global__ void build_ru_kernel(const float* __restrict__ ru, const float* __restrict__ R,
                                bf16* __restrict__ out) {
  int n = blockIdx.x * 256 + threadIdx.x;
  if (n >= NRES) return;
  const float* u = ru + (size_t)n * RDIM;
  const float* r = R + n * 9;
  bf16* o = out + (size_t)n * 128;
  for (int c = 0; c < 32; ++c) o[c] = (bf16)u[c];
  for (int blk = 0; blk < 2; ++blk) {
    int off = 32 + blk * 24;
    for (int v = 0; v < 8; ++v)
      for (int i = 0; i < 3; ++i) {
        float s = r[0 * 3 + i] * u[off + v * 3 + 0] + r[1 * 3 + i] * u[off + v * 3 + 1] +
                  r[2 * 3 + i] * u[off + v * 3 + 2];
        o[off + v * 3 + i] = (bf16)s;
      }
  }
  for (int c = 0; c < 32; ++c) o[80 + c] = (bf16)u[80 + c];
  for (int c = 112; c < 128; ++c) o[c] = (bf16)0.f;
}

// wave-per-row LayerNorm over 128 cols: out = LN(a + resid + bias)*g + b
__global__ void ln_rows_kernel(const float* __restrict__ a, const float* __restrict__ resid,
                               const float* __restrict__ bias, const float* __restrict__ g,
                               const float* __restrict__ b, float* __restrict__ of,
                               bf16* __restrict__ ob) {
  int row = blockIdx.x, lane = threadIdx.x;
  float v[4];
  for (int k = 0; k < 4; ++k) {
    int c = lane * 4 + k;
    float x = a[(size_t)row * 128 + c];
    if (resid) x += resid[(size_t)row * 128 + c];
    if (bias) x += bias[c];
    v[k] = x;
  }
  float s = v[0] + v[1] + v[2] + v[3];
  for (int d = 16; d; d >>= 1) s += __shfl_xor(s, d, 32);
  float mu = s * (1.f / 128.f);
  float var = 0.f;
  for (int k = 0; k < 4; ++k) { float dd = v[k] - mu; var += dd * dd; }
  for (int d = 16; d; d >>= 1) var += __shfl_xor(var, d, 32);
  float rstd = rsqrtf(var * (1.f / 128.f) + 1e-5f);
  for (int k = 0; k < 4; ++k) {
    int c = lane * 4 + k;
    float o = (v[k] - mu) * rstd * g[c] + b[c];
    if (of) of[(size_t)row * 128 + c] = o;
    if (ob) ob[(size_t)row * 128 + c] = (bf16)o;
  }
}

// apply frame to points: out = R p + t ; in/out (NRES, 16*P*3)
__global__ void points_global_kernel(const float* __restrict__ in, const float* __restrict__ R,
                                     const float* __restrict__ tv, float* __restrict__ out, int P) {
  int idx = blockIdx.x * 256 + threadIdx.x;
  int total = NRES * 16 * P;
  if (idx >= total) return;
  int n = idx / (16 * P), hp = idx % (16 * P);
  const float* r = R + n * 9;
  const float* t = tv + n * 3;
  const float* pi = in + (size_t)n * (16 * P * 3) + hp * 3;
  float* po = out + (size_t)n * (16 * P * 3) + hp * 3;
  for (int i = 0; i < 3; ++i)
    po[i] = r[i * 3 + 0] * pi[0] + r[i * 3 + 1] * pi[1] + r[i * 3 + 2] * pi[2] + t[i];
}
__global__ void point_norms_kernel(const float* __restrict__ g, float* __restrict__ out, int P) {
  int idx = blockIdx.x * 256 + threadIdx.x;
  if (idx >= NRES * 16) return;
  int n = idx / 16, h = idx % 16;
  float s = 0.f;
  for (int p = 0; p < P; ++p)
    for (int c = 0; c < 3; ++c) {
      float v = g[(size_t)n * (16 * P * 3) + (h * P + p) * 3 + c];
      s += v * v;
    }
  out[idx] = s;
}
__global__ void softplus16_kernel(const float* __restrict__ w, float* __restrict__ g) {
  int h = threadIdx.x;
  if (h < 16) g[h] = log1pf(expf(w[h]));
}

// logits[h][i][j]
__global__ void logits_kernel(const float* __restrict__ q, const float* __restrict__ k,
                              const float* __restrict__ qpg, const float* __restrict__ kpg,
                              const float* __restrict__ qn, const float* __restrict__ kn,
                              const float* __restrict__ bb, const float* __restrict__ gamma,
                              const float* __restrict__ mask, float* __restrict__ logits) {
  int i = blockIdx.x, h = blockIdx.y;
  __shared__ float qv[16], qp[24], qns, gm, mi;
  int t = threadIdx.x;
  if (t < 16) qv[t] = q[(size_t)i * 256 + h * 16 + t];
  if (t < 24) qp[t] = qpg[(size_t)i * 384 + h * 24 + t];
  if (t == 0) { qns = qn[i * 16 + h]; gm = gamma[h]; mi = mask[i]; }
  __syncthreads();
  const float wl = 0.57735026919f, wc = 1.f / 6.f;
  for (int j = t; j < NRES; j += 128) {
    float qk = 0.f;
    for (int c = 0; c < 16; ++c) qk += qv[c] * k[(size_t)j * 256 + h * 16 + c];
    qk *= 0.25f;
    float dp = 0.f;
    for (int p = 0; p < 24; ++p) dp += qp[p] * kpg[(size_t)j * 384 + h * 24 + p];
    float d2 = qns + kn[j * 16 + h] - 2.f * dp;
    float m2 = mi * mask[j];
    logits[((size_t)h * NRES + i) * NRES + j] =
        wl * (qk + bb[((size_t)i * NRES + j) * 16 + h]) - wl * wc * 0.5f * gm * d2 + (m2 - 1.f) * 1e5f;
  }
}

__global__ void softmax_rows_kernel(const float* __restrict__ logits, bf16* __restrict__ a) {
  int i = blockIdx.x, h = blockIdx.y, lane = threadIdx.x;
  const float* row = logits + ((size_t)h * NRES + i) * NRES;
  float v[12], mx = -1e30f;
  for (int it = 0; it < 12; ++it) { v[it] = row[lane + it * 32]; mx = fmaxf(mx, v[it]); }
  for (int d = 16; d; d >>= 1) mx = fmaxf(mx, __shfl_xor(mx, d, 32));
  float sm = 0.f;
  for (int it = 0; it < 12; ++it) { v[it] = expf(v[it] - mx); sm += v[it]; }
  for (int d = 16; d; d >>= 1) sm += __shfl_xor(sm, d, 32);
  float inv = 1.f / sm;
  bf16* ar = a + ((size_t)h * NRES + i) * NRES;
  for (int it = 0; it < 12; ++it) ar[lane + it * 32] = (bf16)(v[it] * inv);
}

// opair: per residue i, (16h x 384j) @ (384j x 128z) -> cat[:,1024..3071]
__global__ void opair_kernel(const bf16* __restrict__ a, const bf16* __restrict__ zb,
                             float* __restrict__ cat) {
  int i = blockIdx.x;
  int lane = threadIdx.x & 31, w = threadIdx.x >> 5;
  int off = (lane < 16) ? 0 : 16;
  const bf16* arow = a + (size_t)(lane & 15) * NPAIR + (size_t)i * NRES;  // row = head
  const bf16* zi = zb + (size_t)i * (NRES * CZ);
  v8f acc[2] = {{}, {}};
  for (int kt = 0; kt < 12; ++kt) {
    FragU ua;
    const bf16* pa = arow + kt * 32 + off;
    ua.q[0] = *(const uint4*)pa;
    ua.q[1] = *(const uint4*)(pa + 8);
    int kb = kt * 32 + ((lane < 16) ? 0 : 16);
    for (int s = 0; s < 2; ++s) {
      int n0 = (2 * w + s) * 16;
      v16bf bv;
      for (int e = 0; e < 16; ++e) bv[e] = zi[(size_t)(kb + e) * CZ + n0 + (lane & 15)];
      acc[s] = wmma_bf(ua.v, bv, acc[s]);
    }
  }
  for (int s = 0; s < 2; ++s) {
    int col = (2 * w + s) * 16 + (lane & 15);
    for (int r = 0; r < 8; ++r) {
      int h = r + ((lane >> 4) << 3);
      cat[(size_t)i * CATD + 1024 + h * 128 + col] = acc[s][r];
    }
  }
}

// o / opl / nrm -> cat[:,0..1023]
__global__ void ipa_post_kernel(const float* __restrict__ Oapp, const float* __restrict__ R,
                                const float* __restrict__ tv, float* __restrict__ cat) {
  int idx = blockIdx.x * 256 + threadIdx.x;
  if (idx >= NRES * 16) return;
  int i = idx >> 4, h = idx & 15;
  const float* o = Oapp + ((size_t)h * NRES + i) * 64;
  float* crow = cat + (size_t)i * CATD;
  for (int c = 0; c < 16; ++c) crow[h * 16 + c] = o[c];
  const float* r = R + i * 9;
  const float* t = tv + i * 3;
  for (int p = 0; p < PVV; ++p) {
    float d0 = o[16 + p * 3 + 0] - t[0], d1 = o[16 + p * 3 + 1] - t[1], d2 = o[16 + p * 3 + 2] - t[2];
    float l0 = r[0] * d0 + r[3] * d1 + r[6] * d2;  // R^T d
    float l1 = r[1] * d0 + r[4] * d1 + r[7] * d2;
    float l2 = r[2] * d0 + r[5] * d1 + r[8] * d2;
    crow[256 + (h * PVV + p) * 3 + 0] = l0;
    crow[256 + (h * PVV + p) * 3 + 1] = l1;
    crow[256 + (h * PVV + p) * 3 + 2] = l2;
    crow[832 + h * PVV + p] = sqrtf(l0 * l0 + l1 * l1 + l2 * l2 + 1e-8f);
  }
}

// --------------- fused pair MLP + residual + LayerNorm (dominant) -----------
// block = (i, j-tile of 16); 128 threads / 4 waves; WMMA GEMM1(384->128,relu) + GEMM2(128->128)
// z tile staged global->LDS via CDNA5 async-to-LDS engine when available.
__global__ void pair_kernel(const bf16* __restrict__ zb, const bf16* __restrict__ sb,
                            const float* __restrict__ zf, const bf16* __restrict__ We1p,
                            const float* __restrict__ be1, const bf16* __restrict__ We2p,
                            const float* __restrict__ be2, const float* __restrict__ g,
                            const float* __restrict__ b, float* __restrict__ zout) {
  __shared__ bf16 Ap[16 * 384];
  __shared__ bf16 hid[16 * 128];
  __shared__ float ct[16 * 128];
  __shared__ float mu[16], rs[16];
  int i = blockIdx.x, j0 = blockIdx.y * 16;
  int t = threadIdx.x;
#if USE_ASYNC_LDS
  // 16 rows x 256B z tile: 256 b128 transfers, async global->LDS (ASYNCcnt).
  for (int it = 0; it < 2; ++it) {
    int lin = it * 128 + t;                 // 0..255
    int r = lin >> 4, cq = (lin & 15) * 8;  // 8 bf16 = 16B per transfer
    const bf16* gp = &zb[((size_t)i * NRES + j0 + r) * CZ + cq];
    bf16* lp = &Ap[r * 384 + cq];
    __builtin_amdgcn_global_load_async_to_lds_b128(
        (async_gptr)(size_t)gp, (async_lptr)(size_t)lp, 0, 0);
  }
#else
  for (int it = 0; it < 16; ++it) {
    int idx = it * 128 + t;  // 2048 = 16 rows * 128 cols
    int m = idx >> 7, c = idx & 127;
    Ap[m * 384 + c] = zb[((size_t)i * NRES + j0 + m) * CZ + c];
  }
#endif
  // s_i / s_j portions (gathered; stays on the normal path)
  for (int it = 0; it < 32; ++it) {
    int idx = it * 128 + t;  // 4096 = 16 rows * 256 cols
    int m = idx >> 8, c = (idx & 255) + 128;
    bf16 v = (c < 256) ? sb[(size_t)i * CS + (c - 128)] : sb[(size_t)(j0 + m) * CS + (c - 256)];
    Ap[m * 384 + c] = v;
  }
#if USE_ASYNC_LDS
  __builtin_amdgcn_s_wait_asynccnt(0);  // barrier does not cover ASYNCcnt
#endif
  __syncthreads();
  int lane = t & 31, w = t >> 5;
  int off = (lane < 16) ? 0 : 16;
  v8f acc[2] = {{}, {}};
  for (int kt = 0; kt < 12; ++kt) {
    FragU ua;
    const bf16* pa = &Ap[(lane & 15) * 384 + kt * 32 + off];
    ua.q[0] = *(const uint4*)pa;
    ua.q[1] = *(const uint4*)(pa + 8);
    for (int s = 0; s < 2; ++s) {
      int nt = 2 * w + s;
      const bf16* pb = We1p + (size_t)(kt * 8 + nt) * 512 + (size_t)lane * 16;
      __builtin_prefetch(pb + 4096, 0, 1);
      FragU ub;
      ub.q[0] = *(const uint4*)pb;
      ub.q[1] = *(const uint4*)(pb + 8);
      acc[s] = wmma_bf(ua.v, ub.v, acc[s]);
    }
  }
  for (int s = 0; s < 2; ++s) {
    int col = (2 * w + s) * 16 + (lane & 15);
    float bs = be1[col];
    for (int r = 0; r < 8; ++r) {
      int row = r + ((lane >> 4) << 3);
      hid[row * 128 + col] = (bf16)fmaxf(acc[s][r] + bs, 0.f);
    }
  }
  __syncthreads();
  acc[0] = (v8f){};
  acc[1] = (v8f){};
  for (int kt = 0; kt < 4; ++kt) {
    FragU ua;
    const bf16* pa = &hid[(lane & 15) * 128 + kt * 32 + off];
    ua.q[0] = *(const uint4*)pa;
    ua.q[1] = *(const uint4*)(pa + 8);
    for (int s = 0; s < 2; ++s) {
      int nt = 2 * w + s;
      const bf16* pb = We2p + (size_t)(kt * 8 + nt) * 512 + (size_t)lane * 16;
      FragU ub;
      ub.q[0] = *(const uint4*)pb;
      ub.q[1] = *(const uint4*)(pb + 8);
      acc[s] = wmma_bf(ua.v, ub.v, acc[s]);
    }
  }
  for (int s = 0; s < 2; ++s) {
    int col = (2 * w + s) * 16 + (lane & 15);
    float bs = be2[col];
    for (int r = 0; r < 8; ++r) {
      int row = r + ((lane >> 4) << 3);
      ct[row * 128 + col] = acc[s][r] + bs + zf[((size_t)i * NRES + j0 + row) * CZ + col];
    }
  }
  __syncthreads();
  if (t < 16) {
    float s = 0.f;
    for (int c = 0; c < 128; ++c) s += ct[t * 128 + c];
    float m = s * (1.f / 128.f), v = 0.f;
    for (int c = 0; c < 128; ++c) { float d = ct[t * 128 + c] - m; v += d * d; }
    mu[t] = m;
    rs[t] = rsqrtf(v * (1.f / 128.f) + 1e-5f);
  }
  __syncthreads();
  for (int it = 0; it < 16; ++it) {
    int idx = it * 128 + t;
    int row = idx >> 7, col = idx & 127;
    zout[((size_t)i * NRES + j0 + row) * CZ + col] = (ct[idx] - mu[row]) * rs[row] * g[col] + b[col];
  }
}

// ------------------------------- input map ----------------------------------
// Assumes: top-level setup_inputs() insertion order; the `params` pytree is
// flattened jax-style (dict keys sorted, uppercase before lowercase, nested
// tp dicts sorted: so_s,so_v,ss,sv,ve_dot,ve_s,ve_x,vs,vv_s,vv_x).
enum InIdx {
  IN_ATOM_F = 0, IN_BOND_F, IN_RAD_F, IN_EDGE_SH, IN_RES_F, IN_AGG_F, IN_AGG_SH,
  IN_QUATS, IN_TRANS, IN_Z, IN_MASK,
  P_Wbz, P_We1, P_We2, P_Wk, P_Wkp, P_Wout, P_Wq, P_Wqp, P_Wv, P_Wvp,
  P_agg_W1, P_agg_b1,
  PA_so_s, PA_so_v, PA_ss, PA_sv, PA_ve_dot, PA_ve_s, PA_ve_x, PA_vs, PA_vv_s, PA_vv_x,
  PT_so_s, PT_so_v, PT_ss, PT_sv, PT_ve_dot, PT_ve_s, PT_ve_x, PT_vs, PT_vv_s, PT_vv_x,
  P_be1, P_be2, P_bond_W1, P_bond_b1, P_bout, P_head_w,
  P_i2s_W0, P_i2s_W1, P_i2s_W2, P_i2s_b0, P_i2s_b1, P_i2s_b2,
  P_ln1_b, P_ln1_g, P_ln2_b, P_ln2_g, P_lnz_b, P_lnz_g,
  P_rad_W1, P_rad_b1,
  IN_EDGE_IDX, IN_ARB
};

struct Bump {
  char* p; size_t used;
  void* take(size_t bytes) {
    size_t off = (used + 255) & ~(size_t)255;
    used = off + bytes;
    return p + off;
  }
};
static inline int gemm_blocks(int M, int N) { return (((M >> 4) * (N >> 4)) + 3) >> 2; }

extern "C" void kernel_launch(void* const* d_in, const int* in_sizes, int n_in,
                              void* d_out, int out_size, void* d_ws, size_t ws_size,
                              hipStream_t stream) {
  (void)in_sizes; (void)n_in; (void)out_size;
  auto F = [&](int i) { return (const float*)d_in[i]; };
  const int* eidx = (const int*)d_in[IN_EDGE_IDX];
  const int* src = eidx;
  const int* dst = eidx + ETOT;
  const int* arb = (const int*)d_in[IN_ARB];

  float* out_atom = (float*)d_out;
  float* out_s = out_atom + NATOM * ADIM;
  float* out_z = out_s + NRES * CS;

  Bump ws{(char*)d_ws, 0};
  bf16* z_bf    = (bf16*)ws.take((size_t)NPAIR * CZ * 2);
  float* h_buf  = (float*)ws.take((size_t)ETOT * 16 * 4);
  float* hg_buf = (float*)ws.take((size_t)NATOM * 16 * 4);
  float* sums_a = (float*)ws.take((size_t)NATOM * ADIM * 4);   // zero region start
  float* cnt_a  = (float*)ws.take((size_t)NATOM * 4);
  float* sums_r = (float*)ws.take((size_t)NRES * RDIM * 4);
  float* cnt_r  = (float*)ws.take((size_t)NRES * 4);
  size_t zero_bytes = (char*)(cnt_r + NRES) - (char*)sums_a;
  float* res_upd = (float*)ws.take((size_t)NRES * RDIM * 4);
  float* Rbuf   = (float*)ws.take((size_t)NRES * 9 * 4);
  bf16* ru_bf   = (bf16*)ws.take((size_t)NRES * 128 * 2);
  bf16* u1_bf   = (bf16*)ws.take((size_t)NRES * 128 * 2);
  bf16* u2_bf   = (bf16*)ws.take((size_t)NRES * 128 * 2);
  float* u3_f   = (float*)ws.take((size_t)NRES * 128 * 4);
  float* s1_f   = (float*)ws.take((size_t)NRES * 128 * 4);
  bf16* s1_bf   = (bf16*)ws.take((size_t)NRES * 128 * 2);
  float* qf     = (float*)ws.take((size_t)NRES * 256 * 4);
  float* kf     = (float*)ws.take((size_t)NRES * 256 * 4);
  float* vf     = (float*)ws.take((size_t)NRES * 256 * 4);
  float* qpf    = (float*)ws.take((size_t)NRES * 384 * 4);
  float* kpf    = (float*)ws.take((size_t)NRES * 384 * 4);
  float* vpf    = (float*)ws.take((size_t)NRES * 576 * 4);
  float* qpg    = (float*)ws.take((size_t)NRES * 384 * 4);
  float* kpg    = (float*)ws.take((size_t)NRES * 384 * 4);
  float* vpg    = (float*)ws.take((size_t)NRES * 576 * 4);
  float* qn     = (float*)ws.take((size_t)NRES * 16 * 4);
  float* kn     = (float*)ws.take((size_t)NRES * 16 * 4);
  float* gam    = (float*)ws.take(64);
  float* bb     = (float*)ws.take((size_t)NPAIR * 16 * 4);
  float* logits = (float*)ws.take((size_t)NHD * NPAIR * 4);
  bf16* a_bf    = (bf16*)ws.take((size_t)NHD * NPAIR * 2);
  bf16* Bap     = (bf16*)ws.take((size_t)NHD * 12 * 4 * 512 * 2);
  float* Oapp   = (float*)ws.take((size_t)NHD * NRES * 64 * 4);
  float* catb   = (float*)ws.take((size_t)NRES * CATD * 4);
  bf16* cat_bf  = (bf16*)ws.take((size_t)NRES * CATD * 2);
  float* ipa_t  = (float*)ws.take((size_t)NRES * 128 * 4);
  bf16* sfin_bf = (bf16*)ws.take((size_t)NRES * 128 * 2);
  bf16* We1p = (bf16*)ws.take((size_t)384 * 128 * 2);
  bf16* We2p = (bf16*)ws.take((size_t)128 * 128 * 2);
  bf16* Wbzp = (bf16*)ws.take((size_t)128 * 16 * 2);
  bf16* Wq_p = (bf16*)ws.take((size_t)128 * 256 * 2);
  bf16* Wk_p = (bf16*)ws.take((size_t)128 * 256 * 2);
  bf16* Wv_p = (bf16*)ws.take((size_t)128 * 256 * 2);
  bf16* Wqpp = (bf16*)ws.take((size_t)128 * 384 * 2);
  bf16* Wkpp = (bf16*)ws.take((size_t)128 * 384 * 2);
  bf16* Wvpp = (bf16*)ws.take((size_t)128 * 576 * 2);
  bf16* Woutp = (bf16*)ws.take((size_t)3072 * 128 * 2);
  bf16* W0p = (bf16*)ws.take((size_t)128 * 128 * 2);
  bf16* W1p = (bf16*)ws.take((size_t)128 * 128 * 2);
  bf16* W2p = (bf16*)ws.take((size_t)128 * 128 * 2);
  (void)ws_size;

  auto packB = [&](const float* B, bf16* out, int Ksrc, int Kpad, int N) {
    int total = (Kpad >> 5) * (N >> 4) * 512;
    pack_b_kernel<<<dim3((total + 255) / 256), dim3(256), 0, stream>>>(B, out, Ksrc, Kpad, N);
  };
  auto G = [&](const bf16* A, int lda, const bf16* Bp, const float* bias, float* C, bf16* Cb,
               int M, int N, int K, int relu, int batch = 1, long long sA = 0, long long sB = 0,
               long long sC = 0) {
    gemm_bf16_wmma<<<dim3(gemm_blocks(M, N), batch), dim3(128), 0, stream>>>(
        A, lda, Bp, bias, C, Cb, M, N, K, relu, sA, sB, sC);
  };

  (void)hipMemsetAsync(sums_a, 0, zero_bytes, stream);

  // weight packing + conversions
  packB(F(P_We1), We1p, 384, 384, 128);
  packB(F(P_We2), We2p, 128, 128, 128);
  packB(F(P_Wbz), Wbzp, 128, 128, 16);
  packB(F(P_Wq), Wq_p, 128, 128, 256);
  packB(F(P_Wk), Wk_p, 128, 128, 256);
  packB(F(P_Wv), Wv_p, 128, 128, 256);
  packB(F(P_Wqp), Wqpp, 128, 128, 384);
  packB(F(P_Wkp), Wkpp, 128, 128, 384);
  packB(F(P_Wvp), Wvpp, 128, 128, 576);
  packB(F(P_Wout), Woutp, 3072, 3072, 128);
  packB(F(P_i2s_W0), W0p, 112, 128, 128);
  packB(F(P_i2s_W1), W1p, 128, 128, 128);
  packB(F(P_i2s_W2), W2p, 128, 128, 128);
  {
    long long nz = (long long)NPAIR * CZ;
    f32_to_bf16_kernel<<<dim3((int)((nz + 255) / 256)), dim3(256), 0, stream>>>(F(IN_Z), z_bf, nz);
  }

  // edge MLPs
  edge_mlp_kernel<<<dim3((EBOND + 255) / 256), dim3(256), 0, stream>>>(F(IN_BOND_F), F(P_bond_W1), F(P_bond_b1), h_buf, EBOND);
  edge_mlp_kernel<<<dim3((ERAD + 255) / 256), dim3(256), 0, stream>>>(F(IN_RAD_F), F(P_rad_W1), F(P_rad_b1), h_buf + (size_t)EBOND * 16, ERAD);
  edge_mlp_kernel<<<dim3((NATOM + 255) / 256), dim3(256), 0, stream>>>(F(IN_AGG_F), F(P_agg_W1), F(P_agg_b1), hg_buf, NATOM);
  quat_rot_kernel<<<dim3(2), dim3(256), 0, stream>>>(F(IN_QUATS), Rbuf);

  // atom TP messages -> atom_out
  tp_msg_kernel<16, 4><<<dim3((ETOT + 255) / 256), dim3(256), 13312 * 4, stream>>>(
      F(IN_ATOM_F), F(IN_EDGE_SH), h_buf, src, dst,
      F(PT_ss), F(PT_sv), F(PT_vs), F(PT_vv_s), F(PT_vv_x), F(PT_ve_s), F(PT_ve_dot), F(PT_ve_x),
      F(PT_so_s), F(PT_so_v), sums_a, cnt_a, ETOT);
  finalize_atom_kernel<<<dim3((NATOM * ADIM + 255) / 256), dim3(256), 0, stream>>>(F(IN_ATOM_F), sums_a, cnt_a, out_atom);

  // agg TP messages -> residues
  tp_msg_kernel<32, 8><<<dim3((NATOM + 255) / 256), dim3(256), 26624 * 4, stream>>>(
      out_atom, F(IN_AGG_SH), hg_buf, nullptr, arb,
      F(PA_ss), F(PA_sv), F(PA_vs), F(PA_vv_s), F(PA_vv_x), F(PA_ve_s), F(PA_ve_dot), F(PA_ve_x),
      F(PA_so_s), F(PA_so_v), sums_r, cnt_r, NATOM);
  finalize_res_kernel<<<dim3((NRES * RDIM + 255) / 256), dim3(256), 0, stream>>>(sums_r, cnt_r, res_upd);

  // residue i2s MLP (WMMA) + LN1
  build_ru_kernel<<<dim3(2), dim3(256), 0, stream>>>(res_upd, Rbuf, ru_bf);
  G(ru_bf, 128, W0p, F(P_i2s_b0), nullptr, u1_bf, NRES, 128, 128, 1);
  G(u1_bf, 128, W1p, F(P_i2s_b1), nullptr, u2_bf, NRES, 128, 128, 1);
  G(u2_bf, 128, W2p, F(P_i2s_b2), u3_f, nullptr, NRES, 128, 128, 0);
  ln_rows_kernel<<<dim3(NRES), dim3(32), 0, stream>>>(u3_f, F(IN_RES_F), nullptr, F(P_ln1_g), F(P_ln1_b), s1_f, s1_bf);

  // IPA projections (WMMA)
  G(s1_bf, 128, Wq_p, nullptr, qf, nullptr, NRES, 256, 128, 0);
  G(s1_bf, 128, Wk_p, nullptr, kf, nullptr, NRES, 256, 128, 0);
  G(s1_bf, 128, Wv_p, nullptr, vf, nullptr, NRES, 256, 128, 0);
  G(s1_bf, 128, Wqpp, nullptr, qpf, nullptr, NRES, 384, 128, 0);
  G(s1_bf, 128, Wkpp, nullptr, kpf, nullptr, NRES, 384, 128, 0);
  G(s1_bf, 128, Wvpp, nullptr, vpf, nullptr, NRES, 576, 128, 0);
  points_global_kernel<<<dim3((NRES * 16 * 8 + 255) / 256), dim3(256), 0, stream>>>(qpf, Rbuf, F(IN_TRANS), qpg, 8);
  points_global_kernel<<<dim3((NRES * 16 * 8 + 255) / 256), dim3(256), 0, stream>>>(kpf, Rbuf, F(IN_TRANS), kpg, 8);
  points_global_kernel<<<dim3((NRES * 16 * 12 + 255) / 256), dim3(256), 0, stream>>>(vpf, Rbuf, F(IN_TRANS), vpg, 12);
  point_norms_kernel<<<dim3((NRES * 16 + 255) / 256), dim3(256), 0, stream>>>(qpg, qn, 8);
  point_norms_kernel<<<dim3((NRES * 16 + 255) / 256), dim3(256), 0, stream>>>(kpg, kn, 8);
  softplus16_kernel<<<dim3(1), dim3(16), 0, stream>>>(F(P_head_w), gam);
  G(z_bf, 128, Wbzp, nullptr, bb, nullptr, NPAIR, 16, 128, 0);

  // attention
  logits_kernel<<<dim3(NRES, NHD), dim3(128), 0, stream>>>(qf, kf, qpg, kpg, qn, kn, bb, gam, F(IN_MASK), logits);
  softmax_rows_kernel<<<dim3(NRES, NHD), dim3(32), 0, stream>>>(logits, a_bf);
  pack_apply_b<<<dim3((16 * 12 * 4 * 512 + 255) / 256), dim3(256), 0, stream>>>(vf, vpg, Bap);
  G(a_bf, 384, Bap, nullptr, Oapp, nullptr, NRES, 64, NRES, 0, NHD, (long long)NPAIR,
    12 * 4 * 512, (long long)NRES * 64);
  opair_kernel<<<dim3(NRES), dim3(128), 0, stream>>>(a_bf, z_bf, catb);
  ipa_post_kernel<<<dim3((NRES * 16 + 255) / 256), dim3(256), 0, stream>>>(Oapp, Rbuf, F(IN_TRANS), catb);
  {
    long long nc = (long long)NRES * CATD;
    f32_to_bf16_kernel<<<dim3((int)((nc + 255) / 256)), dim3(256), 0, stream>>>(catb, cat_bf, nc);
  }
  G(cat_bf, 3072, Woutp, nullptr, ipa_t, nullptr, NRES, 128, 3072, 0);
  ln_rows_kernel<<<dim3(NRES), dim3(32), 0, stream>>>(ipa_t, s1_f, F(P_bout), F(P_ln2_g), F(P_ln2_b), out_s, sfin_bf);

  // fused pair MLP + LN (dominant stage)
  pair_kernel<<<dim3(NRES, NRES / 16), dim3(128), 0, stream>>>(
      z_bf, sfin_bf, F(IN_Z), We1p, F(P_be1), We2p, F(P_be2), F(P_lnz_g), F(P_lnz_b), out_z);
}
